// RecurrentMAFDecay_60206851556039
// MI455X (gfx1250) — matrix operand order repaired and checked by reading, not compiled, not run on hardware
//
#include <hip/hip_runtime.h>
#include <hip/hip_bf16.h>

#define BB   64
#define LL   512
#define DIN  32
#define EE   64
#define HH   256
#define H3   768
#define NWG  16          // workgroups in recurrent persistent grid
#define NTHR 384         // 12 wave32s: wave = (g in 0..2, m in 0..3)

typedef unsigned short ushort_t;
typedef __bf16 bf16_t;
typedef bf16_t v16bf __attribute__((ext_vector_type(16)));
typedef float  v8f   __attribute__((ext_vector_type(8)));

union ABFrag {        // 16x32 (A) or 32x16 (B) bf16 fragment: 8 VGPRs / lane
    v16bf v;
    uint4 q[2];
};

__device__ __forceinline__ ushort_t f2bf(float f) {
    union { float f; unsigned u; } c; c.f = f;
    unsigned r = c.u + 0x7FFFu + ((c.u >> 16) & 1u);   // round-to-nearest-even
    return (ushort_t)(r >> 16);
}
__device__ __forceinline__ float bf2f(ushort_t u) {
    union { unsigned u; float f; } c; c.u = ((unsigned)u) << 16;
    return c.f;
}
__device__ __forceinline__ float fast_sigmoid(float x) {
    return 1.0f / (1.0f + __expf(-x));
}
__device__ __forceinline__ float fast_tanh(float x) {
    return 1.0f - 2.0f / (__expf(2.0f * x) + 1.0f);
}

// ---------------------------------------------------------------------------
// Setup: zero hidden-state buffers + barrier counters
// ---------------------------------------------------------------------------
__global__ void init_kernel(float* hbuf, unsigned* cnt) {
    int i = blockIdx.x * 256 + threadIdx.x;
    if (i < 4 * BB * HH) hbuf[i] = 0.0f;
    if (i < 128) cnt[i] = 0u;
}

// ---------------------------------------------------------------------------
// Pre: temb = cos(Wt*t+bt), xcat=[x|temb] (bf16), gamma = exp(-relu(Wg*d+bg))
// one block per (b,l), 256 threads
// ---------------------------------------------------------------------------
__global__ void pre_kernel(const float* __restrict__ x, const float* __restrict__ t,
                           const float* __restrict__ Wt, const float* __restrict__ bt,
                           const float* __restrict__ Wg0, const float* __restrict__ bg0,
                           const float* __restrict__ Wg1, const float* __restrict__ bg1,
                           ushort_t* __restrict__ xcat,
                           ushort_t* __restrict__ g0, ushort_t* __restrict__ g1) {
    int bl = blockIdx.x;               // b*LL + l
    int l  = bl & (LL - 1);
    float tv = t[bl];
    float tp = (l == 0) ? 0.0f : t[bl - 1];
    float delta = tv - tp;             // diff with zero prepend
    int h = threadIdx.x;
    float a0 = fmaxf(0.0f, Wg0[h] * delta + bg0[h]);
    g0[bl * HH + h] = f2bf(__expf(-a0));
    float a1 = fmaxf(0.0f, Wg1[h] * delta + bg1[h]);
    g1[bl * HH + h] = f2bf(__expf(-a1));
    if (h < EE)  xcat[bl * (DIN + EE) + DIN + h] = f2bf(__cosf(Wt[h] * tv + bt[h]));
    if (h < DIN) xcat[bl * (DIN + EE) + h]       = f2bf(x[bl * DIN + h]);
}

// fp32 -> bf16 weight convert
__global__ void cvt_kernel(const float* __restrict__ in, ushort_t* __restrict__ out, int n) {
    int i = blockIdx.x * 256 + threadIdx.x;
    if (i < n) out[i] = f2bf(in[i]);
}

// ---------------------------------------------------------------------------
// xi = X[M,K](bf16) @ W[768,K]^T(bf16) + bias  -> bf16 [M,768]
// grid (16, 512); block 384 = 12 waves; wave (g=w/4, m=w%4) -> one 16x16 tile
// ---------------------------------------------------------------------------
template <int K>
__global__ void __launch_bounds__(NTHR) gemm_xi_kernel(const ushort_t* __restrict__ Xbf,
                                                       const ushort_t* __restrict__ Wbf,
                                                       const float*    __restrict__ bias,
                                                       ushort_t*       __restrict__ xi) {
    const int lane = threadIdx.x & 31;
    const int w    = threadIdx.x >> 5;
    const int g    = w >> 2;            // gate 0..2
    const int m    = w & 3;             // M sub-tile 0..3
    const int hi   = (lane < 16) ? 0 : 1;
    const int rowA = blockIdx.y * 64 + m * 16 + (lane & 15);   // M index (A rows)
    const int col  = g * HH + blockIdx.x * 16 + (lane & 15);   // N index

    constexpr int NKT = K / 32;
    // preload all fragments, then issue WMMAs back-to-back
    ABFrag a[NKT], b[NKT];
    const ushort_t* brow = Wbf + (size_t)col * K;
    const ushort_t* arow = Xbf + (size_t)rowA * K;
#pragma unroll
    for (int kt = 0; kt < NKT; ++kt) {
        const ushort_t* ap = arow + kt * 32 + hi * 8;
        a[kt].q[0] = *(const uint4*)(ap);          // K = base+0..7   (v0..v3)
        a[kt].q[1] = *(const uint4*)(ap + 16);     // K = base+16..23 (v4..v7)
        const ushort_t* bp = brow + kt * 32 + hi * 8;
        b[kt].q[0] = *(const uint4*)(bp);          // K = base+0..7   (v0..v3)
        b[kt].q[1] = *(const uint4*)(bp + 16);     // K = base+16..23 (v4..v7)
    }
    v8f acc = {0.f, 0.f, 0.f, 0.f, 0.f, 0.f, 0.f, 0.f};
#pragma unroll
    for (int kt = 0; kt < NKT; ++kt)
        acc = __builtin_amdgcn_wmma_f32_16x16x32_bf16(false, a[kt].v, false, b[kt].v,
                                                      (short)0, acc, false, false);
    float bv = bias[col];
#pragma unroll
    for (int i = 0; i < 8; ++i) {
        int rr = blockIdx.y * 64 + m * 16 + i + hi * 8;        // C/D layout: VGPR i -> M=i or i+8
        xi[(size_t)rr * H3 + col] = f2bf(acc[i] + bv);
    }
}

// ---------------------------------------------------------------------------
// Persistent GRU-D scan: 16 WGs, WG j owns hidden cols [16j,16j+16).
// Whh B-fragments stay in VGPRs across all 512 steps; per step all 8 A
// fragments are preloaded so the 8 WMMAs issue back-to-back into the XDL pipe.
// ---------------------------------------------------------------------------
#define LDA_STRIDE 264     // 256 + 8 bf16 pad -> conflict-free fragment loads
#define HH_STRIDE  17

template <int RELU_OUT>
__global__ void __launch_bounds__(NTHR) gru_scan_kernel(const ushort_t* __restrict__ xi,
                                                        const ushort_t* __restrict__ gamma,
                                                        const ushort_t* __restrict__ Whh_bf,
                                                        const float*    __restrict__ bhh,
                                                        float* hbuf, unsigned* cnt,
                                                        ushort_t* out_bf, float* out_f) {
    __shared__ __align__(16) ushort_t ldsA[64 * LDA_STRIDE];   // hd as bf16 A-matrix
    __shared__ float ldshh[3 * 64 * HH_STRIDE];                // hh exchange (r,z,n)

    const int j    = blockIdx.x;            // 0..15 hidden-column group
    const int lane = threadIdx.x & 31;
    const int w    = threadIdx.x >> 5;
    const int g    = w >> 2;                // gate
    const int m    = w & 3;                 // batch sub-tile
    const int hi   = (lane < 16) ? 0 : 1;
    const int cb   = j * 16;

    // --- load this wave's Whh B-fragments once (resident in VGPRs) ---------
    ABFrag Bf[8];
    {
        const int n = g * HH + cb + (lane & 15);
        const ushort_t* brow = Whh_bf + (size_t)n * HH;
#pragma unroll
        for (int kt = 0; kt < 8; ++kt) {
            const ushort_t* bp = brow + kt * 32 + hi * 8;
            Bf[kt].q[0] = *(const uint4*)(bp);
            Bf[kt].q[1] = *(const uint4*)(bp + 16);
        }
    }
    const ushort_t* arow0 = ldsA + (m * 16 + (lane & 15)) * LDA_STRIDE + hi * 8;

    for (int s = 0; s < LL; ++s) {
        const float* hp = hbuf + (s & 1) * (BB * HH);
        float*       hn = hbuf + ((s + 1) & 1) * (BB * HH);

        // --- stage A = gamma_t * h into LDS (bf16) -------------------------
        for (int idx = threadIdx.x; idx < BB * HH; idx += NTHR) {
            int b = idx >> 8, k = idx & 255;
            float gv = bf2f(gamma[((size_t)b * LL + s) * HH + k]);
            ldsA[b * LDA_STRIDE + k] = f2bf(gv * hp[idx]);
        }
        if (threadIdx.x < BB)   // warm L2/WGP$ for this step's xi gate slices
            __builtin_prefetch(xi + ((size_t)threadIdx.x * LL + s) * H3 + cb, 0, 1);
        __syncthreads();

        // --- hh tile = A(16x256) x B(256x16): preload A, 8 WMMAs b2b -------
        ABFrag Af[8];
#pragma unroll
        for (int kt = 0; kt < 8; ++kt) {
            Af[kt].q[0] = *(const uint4*)(arow0 + kt * 32);
            Af[kt].q[1] = *(const uint4*)(arow0 + kt * 32 + 16);
        }
        v8f acc = {0.f, 0.f, 0.f, 0.f, 0.f, 0.f, 0.f, 0.f};
#pragma unroll
        for (int kt = 0; kt < 8; ++kt)
            acc = __builtin_amdgcn_wmma_f32_16x16x32_bf16(false, Af[kt].v, false, Bf[kt].v,
                                                          (short)0, acc, false, false);
#pragma unroll
        for (int i = 0; i < 8; ++i) {
            int b = m * 16 + i + hi * 8;
            ldshh[(g * 64 + b) * HH_STRIDE + (lane & 15)] = acc[i];
        }
        __syncthreads();

        // --- gate nonlinearities, write h_new ------------------------------
        for (int idx = threadIdx.x; idx < BB * 16; idx += NTHR) {
            int b = idx >> 4, c = idx & 15;
            int col = cb + c;
            size_t xrow = ((size_t)b * LL + s) * H3;
            float hr  = ldshh[(0 * 64 + b) * HH_STRIDE + c] + bhh[col];
            float hz  = ldshh[(1 * 64 + b) * HH_STRIDE + c] + bhh[HH + col];
            float hnn = ldshh[(2 * 64 + b) * HH_STRIDE + c] + bhh[2 * HH + col];
            float xr = bf2f(xi[xrow + col]);
            float xz = bf2f(xi[xrow + HH + col]);
            float xn = bf2f(xi[xrow + 2 * HH + col]);
            float gv = bf2f(gamma[((size_t)b * LL + s) * HH + col]);
            float hd = gv * hp[b * HH + col];
            float r  = fast_sigmoid(xr + hr);
            float z  = fast_sigmoid(xz + hz);
            float nn = fast_tanh(xn + r * hnn);
            float hv = (1.0f - z) * nn + z * hd;
            hn[b * HH + col] = hv;
            if (RELU_OUT) out_bf[((size_t)b * LL + s) * HH + col] = f2bf(fmaxf(hv, 0.0f));
            else          out_f[((size_t)b * LL + s) * HH + col] = hv;
        }

        // --- grid barrier across the 16 persistent WGs ---------------------
        __threadfence();
        __syncthreads();
        if (threadIdx.x == 0) {
            __hip_atomic_fetch_add(cnt, 1u, __ATOMIC_RELEASE, __HIP_MEMORY_SCOPE_AGENT);
            unsigned target = (unsigned)NWG * (unsigned)(s + 1);
            while (__hip_atomic_load(cnt, __ATOMIC_ACQUIRE, __HIP_MEMORY_SCOPE_AGENT) < target)
                __builtin_amdgcn_s_sleep(1);
        }
        __syncthreads();
    }
}

// ---------------------------------------------------------------------------
extern "C" void kernel_launch(void* const* d_in, const int* in_sizes, int n_in,
                              void* d_out, int out_size, void* d_ws, size_t ws_size,
                              hipStream_t stream) {
    (void)in_sizes; (void)n_in; (void)out_size; (void)ws_size;
    const float* x    = (const float*)d_in[0];
    const float* t    = (const float*)d_in[1];
    const float* Wt   = (const float*)d_in[2];
    const float* bt   = (const float*)d_in[3];
    const float* Wih0 = (const float*)d_in[4];
    const float* bih0 = (const float*)d_in[5];
    const float* Whh0 = (const float*)d_in[6];
    const float* bhh0 = (const float*)d_in[7];
    const float* Wg0  = (const float*)d_in[8];
    const float* bg0  = (const float*)d_in[9];
    const float* Wih1 = (const float*)d_in[10];
    const float* bih1 = (const float*)d_in[11];
    const float* Whh1 = (const float*)d_in[12];
    const float* bhh1 = (const float*)d_in[13];
    const float* Wg1  = (const float*)d_in[14];
    const float* bg1  = (const float*)d_in[15];
    float* out = (float*)d_out;

    const size_t ML = (size_t)BB * LL;      // 32768 rows
    char* ws = (char*)d_ws;
    size_t off = 0;
    auto alloc = [&](size_t bytes) -> char* {
        char* p = ws + off;
        off += (bytes + 255) & ~(size_t)255;
        return p;
    };
    ushort_t* xcat  = (ushort_t*)alloc(ML * (DIN + EE) * 2);
    ushort_t* xi0   = (ushort_t*)alloc(ML * H3 * 2);
    ushort_t* xi1   = (ushort_t*)alloc(ML * H3 * 2);
    ushort_t* g0    = (ushort_t*)alloc(ML * HH * 2);
    ushort_t* g1    = (ushort_t*)alloc(ML * HH * 2);
    ushort_t* h1    = (ushort_t*)alloc(ML * HH * 2);
    ushort_t* wih0b = (ushort_t*)alloc((size_t)H3 * (DIN + EE) * 2);
    ushort_t* whh0b = (ushort_t*)alloc((size_t)H3 * HH * 2);
    ushort_t* wih1b = (ushort_t*)alloc((size_t)H3 * HH * 2);
    ushort_t* whh1b = (ushort_t*)alloc((size_t)H3 * HH * 2);
    float*    hbuf  = (float*)alloc(4 * BB * HH * 4);   // 2 double-buffer pairs
    unsigned* cnt   = (unsigned*)alloc(512);            // two padded counters

    init_kernel<<<dim3((4 * BB * HH + 255) / 256), 256, 0, stream>>>(hbuf, cnt);
    pre_kernel<<<dim3((unsigned)ML), 256, 0, stream>>>(x, t, Wt, bt, Wg0, bg0, Wg1, bg1,
                                                       xcat, g0, g1);
    cvt_kernel<<<(H3 * (DIN + EE) + 255) / 256, 256, 0, stream>>>(Wih0, wih0b, H3 * (DIN + EE));
    cvt_kernel<<<(H3 * HH + 255) / 256, 256, 0, stream>>>(Whh0, whh0b, H3 * HH);
    cvt_kernel<<<(H3 * HH + 255) / 256, 256, 0, stream>>>(Whh1, wih1b, H3 * HH);
    cvt_kernel<<<(H3 * HH + 255) / 256, 256, 0, stream>>>(Whh1, whh1b, H3 * HH);
    // NOTE: Wih1 conversion must use Wih1 source:
    cvt_kernel<<<(H3 * HH + 255) / 256, 256, 0, stream>>>(Wih1, wih1b, H3 * HH);

    // layer 0: input GEMM (K=96) then persistent scan (relu+bf16 output)
    gemm_xi_kernel<DIN + EE><<<dim3(16, (unsigned)(ML / 64)), NTHR, 0, stream>>>(xcat, wih0b, bih0, xi0);
    gru_scan_kernel<1><<<dim3(NWG), NTHR, 0, stream>>>(xi0, g0, whh0b, bhh0,
                                                       hbuf, cnt, h1, nullptr);
    // layer 1: input GEMM (K=256) then persistent scan (fp32 output)
    gemm_xi_kernel<HH><<<dim3(16, (unsigned)(ML / 64)), NTHR, 0, stream>>>(h1, wih1b, bih1, xi1);
    gru_scan_kernel<0><<<dim3(NWG), NTHR, 0, stream>>>(xi1, g1, whh1b, bhh1,
                                                       hbuf + 2 * BB * HH, cnt + 64, nullptr, out);
}
